// AtomAttentionEncoder_11871289606514
// MI455X (gfx1250) — matrix operand order repaired and checked by reading, not compiled, not run on hardware
//
#include <hip/hip_runtime.h>

typedef __bf16 bf16;
typedef __attribute__((ext_vector_type(8)))  bf16  v8bf;
typedef __attribute__((ext_vector_type(16))) bf16  v16bf;
typedef __attribute__((ext_vector_type(8)))  float v8f;

#define N_ATOM  8192
#define N_TOKEN 512
#define C_ATOM  128
#define C_Z     128
#define C_PAIR  16
#define C_TOKEN 384
#define NQ      32
#define NK      128
#define NH      4
#define DH      32
#define NBQ     256   // N_ATOM / NQ
#define APT     16    // atoms per token
#define NPAIR   (NBQ * NQ * NK)
#define FEAT    389
#define FPAD    416   // 13 * 32

// ---------------------------------------------------------------- device utils
__device__ __forceinline__ float sigm(float x) { return 1.f / (1.f + __expf(-x)); }

// A-fragment for v_wmma_f32_16x16x32_bf16 (16x32, row-major source).
// Caller passes pointer already offset to row*lda + k0 + half*16.
// Layout: lanes 0-15 hold K {0..15}, lanes 16-31 hold K {16..31} (matches B).
__device__ __forceinline__ v16bf load_fragA(const bf16* p) {
  v8bf lo = *(const v8bf*)(p);        // K = k0 + half*16 + 0..7
  v8bf hi = *(const v8bf*)(p + 8);    // K = k0 + half*16 + 8..15
  v16bf f;
#pragma unroll
  for (int i = 0; i < 8; ++i) { f[i] = lo[i]; f[8 + i] = hi[i]; }
  return f;
}

__device__ __forceinline__ v8f wmma_bf16(v16bf a, v16bf b, v8f c) {
  return __builtin_amdgcn_wmma_f32_16x16x32_bf16(false, a, false, b, (short)0, c,
                                                 false, false);
}

// ---------------------------------------------------------------- WMMA GEMM, 32x(16*NT)/wave
// C[M,N] = act(A[M,K] @ B[K,N] (+ Cadd)); A bf16 row-major (lda); B transposed bf16
// Bt[N][K] (ldb).  Each wave owns a 32 x (16*NT) macro tile: 2 A-frags x NT B-frags
// -> 2*NT WMMAs per K-step, doubling operand reuse vs a single 16x16 tile.
template <int NT>
__global__ __launch_bounds__(256) void k_gemm2(
    const bf16* __restrict__ A, int lda,
    const bf16* __restrict__ Bt, int ldb,
    float* __restrict__ Cf, bf16* __restrict__ Cb,
    const float* __restrict__ Cadd,
    int M, int N, int K, int ldc, int ldcb, int act)
{
  int job   = blockIdx.x * 8 + (threadIdx.x >> 5);
  int jobsN = N / (16 * NT);
  int jobs  = (M >> 5) * jobsN;
  if (job >= jobs) return;
  int lane = threadIdx.x & 31;
  int half = lane >> 4, l15 = lane & 15;
  int jm = job / jobsN, jn = job % jobsN;
  const bf16* arow0 = A + (size_t)(jm * 32 + l15) * lda + half * 16;
  const bf16* arow1 = arow0 + (size_t)16 * lda;
  const bf16* brow[NT];
#pragma unroll
  for (int i = 0; i < NT; ++i)
    brow[i] = Bt + (size_t)(jn * 16 * NT + i * 16 + l15) * ldb + half * 16;
  v8f zero = {};
  v8f acc[2][NT];
#pragma unroll
  for (int x = 0; x < 2; ++x)
#pragma unroll
    for (int i = 0; i < NT; ++i) acc[x][i] = zero;
  for (int k0 = 0; k0 < K; k0 += 32) {
    v16bf a0 = load_fragA(arow0 + k0);
    v16bf a1 = load_fragA(arow1 + k0);
#pragma unroll
    for (int i = 0; i < NT; ++i) {
      v16bf b = *(const v16bf*)(brow[i] + k0);
      acc[0][i] = wmma_bf16(a0, b, acc[0][i]);
      acc[1][i] = wmma_bf16(a1, b, acc[1][i]);
    }
  }
#pragma unroll
  for (int x = 0; x < 2; ++x)
#pragma unroll
    for (int i = 0; i < NT; ++i) {
      int col = jn * 16 * NT + i * 16 + l15;
#pragma unroll
      for (int r = 0; r < 8; ++r) {
        int row = jm * 32 + x * 16 + r + half * 8;
        float v = acc[x][i][r];
        if (Cadd) v += Cadd[(size_t)row * ldc + col];
        if (act)  v = fmaxf(v, 0.f);
        if (Cf) Cf[(size_t)row * ldc + col] = v;
        if (Cb) Cb[(size_t)row * ldcb + col] = (bf16)v;
      }
    }
}

// ---------------------------------------------------------------- row LayerNorm
__global__ void k_ln(const float* __restrict__ X, int D,
                     float* __restrict__ Yf, bf16* __restrict__ Yb, int rows)
{
  int row = blockIdx.x * 4 + (threadIdx.x >> 5);
  if (row >= rows) return;
  int lane = threadIdx.x & 31;
  const float* x = X + (size_t)row * D;
  float s = 0.f, s2 = 0.f;
  for (int i = lane; i < D; i += 32) { float v = x[i]; s += v; s2 += v * v; }
#pragma unroll
  for (int m = 16; m; m >>= 1) { s += __shfl_xor(s, m, 32); s2 += __shfl_xor(s2, m, 32); }
  float mu = s / (float)D;
  float rs = rsqrtf(s2 / (float)D - mu * mu + 1e-5f);
  for (int i = lane; i < D; i += 32) {
    float v = (x[i] - mu) * rs;
    if (Yf) Yf[(size_t)row * D + i] = v;
    if (Yb) Yb[(size_t)row * D + i] = (bf16)v;
  }
}

// ---------------------------------------------------------------- feature packing
// featb[a][0:416] = concat(ref_pos, charge, mask, element, name_chars, zero-pad) as bf16
__global__ void k_packfeat(const float* __restrict__ ref_pos, const float* __restrict__ ref_charge,
                           const float* __restrict__ ref_mask, const float* __restrict__ ref_elem,
                           const float* __restrict__ ref_anc, bf16* __restrict__ featb)
{
  int idx = blockIdx.x * blockDim.x + threadIdx.x;
  if (idx >= N_ATOM * FPAD) return;
  int a = idx / FPAD, f = idx % FPAD;
  float v;
  if (f < 3)        v = ref_pos[a * 3 + f];
  else if (f == 3)  v = ref_charge[a];
  else if (f == 4)  v = ref_mask[a];
  else if (f < 133) v = ref_elem[a * 128 + (f - 5)];
  else if (f < FEAT) v = ref_anc[a * 256 + (f - 133)];
  else              v = 0.f;
  featb[idx] = (bf16)v;
}

// ---------------------------------------------------------------- embedding epilogue
// c = cfeat + s_proj[tok];  q = c + r_l @ w_r;  cr_bf = bf16(relu(c))
__global__ void k_embed2(const float* __restrict__ cfeat, const float* __restrict__ r_l,
                         const float* __restrict__ w_r, const float* __restrict__ s_proj,
                         const int* __restrict__ a2t,
                         float* __restrict__ c_out, float* __restrict__ q_out,
                         bf16* __restrict__ cr_bf)
{
  int idx = blockIdx.x * blockDim.x + threadIdx.x;
  if (idx >= N_ATOM * C_ATOM) return;
  int a = idx >> 7, j = idx & 127;
  float acc = cfeat[idx] + s_proj[(size_t)a2t[a] * C_ATOM + j];
  c_out[idx] = acc;
  cr_bf[idx] = (bf16)fmaxf(acc, 0.f);
  q_out[idx] = acc + r_l[a * 3 + 0] * w_r[0 * C_ATOM + j] +
               r_l[a * 3 + 1] * w_r[1 * C_ATOM + j] +
               r_l[a * 3 + 2] * w_r[2 * C_ATOM + j];
}

// ---------------------------------------------------------------- fused pair stack
// One workgroup per (blk, qi) query row: computes p0 for its 128 keys into LDS,
// runs the 3-layer 16x16 pair MLP entirely in LDS with WMMA (K padded to 32),
// writes only the final p (+p0 residual) to global. No global intermediates.
__global__ __launch_bounds__(256) void k_pair_fused(
    const float* __restrict__ ref_pos, const int* __restrict__ uid,
    const int* __restrict__ a2t, const float* __restrict__ zp,
    const float* __restrict__ cq, const float* __restrict__ ck,
    const float* __restrict__ w_d, const float* __restrict__ w_inv,
    const float* __restrict__ w_v,
    const bf16* __restrict__ m1T, const bf16* __restrict__ m2T,
    const bf16* __restrict__ m3T,
    float* __restrict__ out_p)
{
  __shared__ float sp0[NK][C_PAIR];                          //  8 KB residual
  __shared__ __attribute__((aligned(16))) bf16 sa[NK][32];   //  8 KB ping
  __shared__ __attribute__((aligned(16))) bf16 sb[NK][32];   //  8 KB pong
  int qa  = blockIdx.x;            // atom (query) index
  int blk = qa >> 5;
  int tid = threadIdx.x;
  // ---- p0 for 128 keys x 16 channels (2 threads per key) ----
  {
    int row = tid >> 1;
    int c0  = (tid & 1) * 8;
    int ka  = blk * NQ + NQ / 2 - NK / 2 + row;
    bool valid = (ka >= 0) && (ka < N_ATOM);
    int kc = ka < 0 ? 0 : (ka > N_ATOM - 1 ? N_ATOM - 1 : ka);
    float vm = (valid && (uid[qa] == uid[kc])) ? 1.f : 0.f;
    float dx = ref_pos[qa * 3 + 0] - ref_pos[kc * 3 + 0];
    float dy = ref_pos[qa * 3 + 1] - ref_pos[kc * 3 + 1];
    float dz = ref_pos[qa * 3 + 2] - ref_pos[kc * 3 + 2];
    float inv = 1.f / (1.f + dx * dx + dy * dy + dz * dz);
    const float* zrow = zp + ((size_t)a2t[qa] * N_TOKEN + a2t[kc]) * C_PAIR;
    const float* cqr  = cq + (size_t)qa * C_PAIR;
    const float* ckr  = ck + (size_t)kc * C_PAIR;
#pragma unroll
    for (int jj = 0; jj < 8; ++jj) {
      int j = c0 + jj;
      float v = (dx * w_d[0 * 16 + j] + dy * w_d[1 * 16 + j] + dz * w_d[2 * 16 + j] +
                 inv * w_inv[j] + vm * w_v[j]) * vm;
      v += zrow[j] + cqr[j] + ckr[j];
      sp0[row][j]     = v;
      sa[row][j]      = (bf16)fmaxf(v, 0.f);
      sa[row][16 + j] = (bf16)0.f;      // K pad
      sb[row][16 + j] = (bf16)0.f;      // K pad
    }
  }
  __syncthreads();
  int lane = tid & 31, wave = tid >> 5, half = lane >> 4, l15 = lane & 15;
  // ---- layer 1: relu(sa @ m1) -> sb ----
  {
    v16bf af = load_fragA(&sa[wave * 16 + l15][half * 16]);
    v16bf bf_ = *(const v16bf*)(m1T + (size_t)l15 * 32 + half * 16);
    v8f acc = {};
    acc = wmma_bf16(af, bf_, acc);
#pragma unroll
    for (int r = 0; r < 8; ++r)
      sb[wave * 16 + r + half * 8][l15] = (bf16)fmaxf(acc[r], 0.f);
  }
  __syncthreads();
  // ---- layer 2: relu(sb @ m2) -> sa ----
  {
    v16bf af = load_fragA(&sb[wave * 16 + l15][half * 16]);
    v16bf bf_ = *(const v16bf*)(m2T + (size_t)l15 * 32 + half * 16);
    v8f acc = {};
    acc = wmma_bf16(af, bf_, acc);
#pragma unroll
    for (int r = 0; r < 8; ++r)
      sa[wave * 16 + r + half * 8][l15] = (bf16)fmaxf(acc[r], 0.f);
  }
  __syncthreads();
  // ---- layer 3: p = p0 + sa @ m3 -> global ----
  {
    v16bf af = load_fragA(&sa[wave * 16 + l15][half * 16]);
    v16bf bf_ = *(const v16bf*)(m3T + (size_t)l15 * 32 + half * 16);
    v8f acc = {};
    acc = wmma_bf16(af, bf_, acc);
#pragma unroll
    for (int r = 0; r < 8; ++r) {
      int row = wave * 16 + r + half * 8;
      out_p[((size_t)qa * NK + row) * C_PAIR + l15] = acc[r] + sp0[row][l15];
    }
  }
}

// ---------------------------------------------------------------- bias for all 3 blocks
// biasq[b] = ln(p) @ w_bias[b] + neg   (single pass over the 64 MB p tensor)
__global__ void k_bias3(const float* __restrict__ p,
                        const float* __restrict__ wb0, const float* __restrict__ wb1,
                        const float* __restrict__ wb2,
                        float* __restrict__ bq0, float* __restrict__ bq1,
                        float* __restrict__ bq2)
{
  int idx = blockIdx.x * blockDim.x + threadIdx.x;
  if (idx >= NPAIR) return;
  int ki  = idx & (NK - 1);
  int blk = idx >> 12;
  const float* pr = p + (size_t)idx * C_PAIR;
  float v[C_PAIR];
  float s = 0.f, s2 = 0.f;
#pragma unroll
  for (int j = 0; j < C_PAIR; ++j) { v[j] = pr[j]; s += v[j]; s2 += v[j] * v[j]; }
  float mu = s / 16.f;
  float rs = rsqrtf(s2 / 16.f - mu * mu + 1e-5f);
#pragma unroll
  for (int j = 0; j < C_PAIR; ++j) v[j] = (v[j] - mu) * rs;
  int ka = blk * NQ + NQ / 2 - NK / 2 + ki;
  float neg = (ka >= 0 && ka < N_ATOM) ? 0.f : -1e9f;
#pragma unroll
  for (int h = 0; h < NH; ++h) {
    float b0 = neg, b1 = neg, b2 = neg;
#pragma unroll
    for (int j = 0; j < C_PAIR; ++j) {
      b0 += v[j] * wb0[j * NH + h];
      b1 += v[j] * wb1[j * NH + h];
      b2 += v[j] * wb2[j * NH + h];
    }
    bq0[(size_t)idx * NH + h] = b0;
    bq1[(size_t)idx * NH + h] = b1;
    bq2[(size_t)idx * NH + h] = b2;
  }
}

// ---------------------------------------------------------------- AdaLN
__global__ void k_adaln(const float* __restrict__ q, const float* __restrict__ G1,
                        const float* __restrict__ G2, bf16* __restrict__ b_bf)
{
  int row = blockIdx.x * 4 + (threadIdx.x >> 5);
  if (row >= N_ATOM) return;
  int lane = threadIdx.x & 31;
  const float* x = q + (size_t)row * C_ATOM;
  float s = 0.f, s2 = 0.f;
#pragma unroll
  for (int t = 0; t < 4; ++t) { float v = x[lane + 32 * t]; s += v; s2 += v * v; }
#pragma unroll
  for (int m = 16; m; m >>= 1) { s += __shfl_xor(s, m, 32); s2 += __shfl_xor(s2, m, 32); }
  float mu = s / 128.f;
  float rs = rsqrtf(s2 / 128.f - mu * mu + 1e-5f);
#pragma unroll
  for (int t = 0; t < 4; ++t) {
    int i = lane + 32 * t;
    size_t o = (size_t)row * C_ATOM + i;
    float v = (x[i] - mu) * rs;
    b_bf[o] = (bf16)(v * sigm(G1[o]) + G2[o]);
  }
}

// ---------------------------------------------------------------- windowed attention
// One workgroup per 32x128 window, looping over the 4 heads. Logits: one WMMA per
// 16x16 tile (K=DH=32). Softmax staged in LDS. v-tile staged transposed in LDS so
// the o = w @ v WMMA reads both operands via contiguous ds loads.
__global__ __launch_bounds__(256) void k_attn(
    const bf16* __restrict__ qh, const bf16* __restrict__ kh,
    const bf16* __restrict__ vh, const float* __restrict__ biasq,
    float* __restrict__ obuf)
{
  __shared__ float slog[NQ][NK];                                   // 16 KB
  __shared__ __attribute__((aligned(16))) bf16 sw[NQ][NK];         //  8 KB
  __shared__ __attribute__((aligned(16))) bf16 svt[DH][NK];        //  8 KB
  int blk  = blockIdx.x;
  int tid  = threadIdx.x, lane = tid & 31, wave = tid >> 5;
  int half = lane >> 4, l15 = lane & 15;
  int start = blk * NQ + NQ / 2 - NK / 2;
  const float isq = 0.17677669529663687f;  // 1/sqrt(DH)
  for (int h = 0; h < NH; ++h) {
    // ---- logits: 16 tiles (2 x 8), 2 per wave ----
#pragma unroll
    for (int tt = 0; tt < 2; ++tt) {
      int t  = wave * 2 + tt;
      int tm = t >> 3, tn = t & 7;
      int qa = blk * NQ + tm * 16 + l15;
      v16bf af = load_fragA(qh + (size_t)qa * C_ATOM + h * DH + half * 16);
      int kk = start + tn * 16 + l15;
      int kc = kk < 0 ? 0 : (kk > N_ATOM - 1 ? N_ATOM - 1 : kk);
      v16bf bfr = *(const v16bf*)(kh + (size_t)kc * C_ATOM + h * DH + half * 16);
      v8f acc = {};
      acc = wmma_bf16(af, bfr, acc);
      int col = tn * 16 + l15;
#pragma unroll
      for (int r = 0; r < 8; ++r) {
        int row = tm * 16 + r + half * 8;
        slog[row][col] = acc[r] * isq +
                         biasq[(((size_t)blk * NQ + row) * NK + col) * NH + h];
      }
    }
    // ---- stage v tile transposed: svt[dh][key] ----
#pragma unroll
    for (int e = tid * 16; e < tid * 16 + 16; ++e) {
      int dh = e >> 7, key = e & 127;
      int kk = start + key;
      int kc = kk < 0 ? 0 : (kk > N_ATOM - 1 ? N_ATOM - 1 : kk);
      svt[dh][key] = vh[(size_t)kc * C_ATOM + h * DH + dh];
    }
    __syncthreads();
    // ---- softmax over keys ----
    if (tid < NQ) {
      float mx = -3.4e38f;
      for (int k2 = 0; k2 < NK; ++k2) mx = fmaxf(mx, slog[tid][k2]);
      float sum = 0.f;
      for (int k2 = 0; k2 < NK; ++k2) {
        float e = __expf(slog[tid][k2] - mx);
        slog[tid][k2] = e; sum += e;
      }
      float rs = 1.f / sum;
      for (int k2 = 0; k2 < NK; ++k2) sw[tid][k2] = (bf16)(slog[tid][k2] * rs);
    }
    __syncthreads();
    // ---- o = w @ v : 4 tiles (2 x 2), waves 0..3, all operands from LDS ----
    if (wave < 4) {
      int tm = wave >> 1, tn = wave & 1;
      v8f acc = {};
      for (int k0 = 0; k0 < NK; k0 += 32) {
        v16bf af  = load_fragA(&sw[tm * 16 + l15][k0 + half * 16]);
        v16bf bfr = *(const v16bf*)(&svt[tn * 16 + l15][k0 + half * 16]);
        acc = wmma_bf16(af, bfr, acc);
      }
      int col = h * DH + tn * 16 + l15;
#pragma unroll
      for (int r = 0; r < 8; ++r) {
        int row = blk * NQ + tm * 16 + r + half * 8;
        obuf[(size_t)row * C_ATOM + col] = acc[r];
      }
    }
    __syncthreads();
  }
}

// ---------------------------------------------------------------- elementwise helpers
__global__ void k_gate(const float* __restrict__ G, const float* __restrict__ o,
                       bf16* __restrict__ og, int n) {
  int i = blockIdx.x * blockDim.x + threadIdx.x;
  if (i < n) og[i] = (bf16)(sigm(G[i]) * o[i]);
}
__global__ void k_skipadd(float* __restrict__ q, const float* __restrict__ G,
                          const float* __restrict__ x, int n) {
  int i = blockIdx.x * blockDim.x + threadIdx.x;
  if (i < n) q[i] += sigm(G[i]) * x[i];
}
__global__ void k_silu2(const float* __restrict__ H1, const float* __restrict__ H2,
                        bf16* __restrict__ hh, int n) {
  int i = blockIdx.x * blockDim.x + threadIdx.x;
  if (i < n) { float a = H1[i]; hh[i] = (bf16)(a * sigm(a) * H2[i]); }
}
__global__ void k_cast(const float* __restrict__ x, bf16* __restrict__ y, int n) {
  int i = blockIdx.x * blockDim.x + threadIdx.x;
  if (i < n) y[i] = (bf16)x[i];
}
__global__ void k_castT(const float* __restrict__ B, bf16* __restrict__ Bt,
                        int K, int N, int Kpad) {
  int i = blockIdx.x * blockDim.x + threadIdx.x;
  if (i >= N * Kpad) return;
  int n = i / Kpad, k = i % Kpad;
  Bt[i] = (k < K) ? (bf16)B[(size_t)k * N + n] : (bf16)0.f;
}
__global__ void k_mean(const float* __restrict__ a, float* __restrict__ out) {
  int i = blockIdx.x * blockDim.x + threadIdx.x;
  if (i >= N_TOKEN * C_TOKEN) return;
  int t = i / C_TOKEN, j = i % C_TOKEN;
  float s = 0.f;
  for (int u = 0; u < APT; ++u) s += a[(size_t)(t * APT + u) * C_TOKEN + j];
  out[i] = s * (1.f / APT);
}

// ---------------------------------------------------------------- host wrappers
static inline void gemm(hipStream_t st, const bf16* A, int lda, const bf16* Bt, int ldb,
                        float* Cf, bf16* Cb, const float* Cadd,
                        int M, int N, int K, int ldc, int ldcb, int act) {
  if ((N & 31) == 0) {
    int jobs = (M / 32) * (N / 32);
    k_gemm2<2><<<(jobs + 7) / 8, 256, 0, st>>>(A, lda, Bt, ldb, Cf, Cb, Cadd,
                                               M, N, K, ldc, ldcb, act);
  } else {
    int jobs = (M / 32) * (N / 16);
    k_gemm2<1><<<(jobs + 7) / 8, 256, 0, st>>>(A, lda, Bt, ldb, Cf, Cb, Cadd,
                                               M, N, K, ldc, ldcb, act);
  }
}
static inline void castT(hipStream_t st, const float* B, bf16* Bt, int K, int N, int Kpad) {
  int n = N * Kpad;
  k_castT<<<(n + 255) / 256, 256, 0, st>>>(B, Bt, K, N, Kpad);
}

extern "C" void kernel_launch(void* const* d_in, const int* in_sizes, int n_in,
                              void* d_out, int out_size, void* d_ws, size_t ws_size,
                              hipStream_t stream)
{
  (void)in_sizes; (void)n_in; (void)out_size; (void)ws_size;
  const float* ref_pos    = (const float*)d_in[0];
  const float* ref_charge = (const float*)d_in[1];
  const float* ref_mask   = (const float*)d_in[2];
  const float* ref_elem   = (const float*)d_in[3];
  const float* ref_anc    = (const float*)d_in[4];
  const float* r_l        = (const float*)d_in[5];
  const float* s_in       = (const float*)d_in[6];
  const float* z_in       = (const float*)d_in[7];
  // params: jax pytree leaves, dict keys sorted; blocks (3 x 15 leaves) then globals
  auto PL = [&](int i) { return (const float*)d_in[8 + i]; };
  const int*   uid        = (const int*)d_in[66];
  const int*   a2t        = (const int*)d_in[67];

  // outputs: a_tok, q, c, p concatenated (f32)
  float* out_atok = (float*)d_out;
  float* out_q    = out_atok + (size_t)N_TOKEN * C_TOKEN;
  float* out_c    = out_q + (size_t)N_ATOM * C_ATOM;
  float* out_p    = out_c + (size_t)N_ATOM * C_ATOM;

  // workspace bump allocator
  char* wsp = (char*)d_ws; size_t off = 0;
  auto alloc = [&](size_t bytes) -> void* {
    void* p = wsp + off; off = (off + bytes + 255) & ~(size_t)255; return p;
  };
  bf16*  s_ln   = (bf16*)alloc((size_t)N_TOKEN * 384 * 2);
  float* s_proj = (float*)alloc((size_t)N_TOKEN * C_ATOM * 4);
  bf16*  featb  = (bf16*)alloc((size_t)N_ATOM * FPAD * 2);
  float* cfeat  = (float*)alloc((size_t)N_ATOM * C_ATOM * 4);
  float* qbuf   = (float*)alloc((size_t)N_ATOM * C_ATOM * 4);
  bf16*  lnc    = (bf16*)alloc((size_t)N_ATOM * C_ATOM * 2);
  bf16*  crb    = (bf16*)alloc((size_t)N_ATOM * C_ATOM * 2);
  float* zp     = (float*)alloc((size_t)N_TOKEN * N_TOKEN * C_PAIR * 4);
  float* cqb    = (float*)alloc((size_t)N_ATOM * C_PAIR * 4);
  float* ckb    = (float*)alloc((size_t)N_ATOM * C_PAIR * 4);
  float* bq3    = (float*)alloc((size_t)3 * NPAIR * NH * 4);   // bias for all 3 blocks
  bf16*  big0   = (bf16*)alloc((size_t)67108864);              // ln(z) bf16
  float* G1     = (float*)alloc((size_t)N_ATOM * C_ATOM * 4);
  float* G2     = (float*)alloc((size_t)N_ATOM * C_ATOM * 4);
  bf16*  bbf    = (bf16*)alloc((size_t)N_ATOM * C_ATOM * 2);
  bf16*  qhb    = (bf16*)alloc((size_t)N_ATOM * C_ATOM * 2);
  bf16*  khb    = (bf16*)alloc((size_t)N_ATOM * C_ATOM * 2);
  bf16*  vhb    = (bf16*)alloc((size_t)N_ATOM * C_ATOM * 2);
  float* obuf   = (float*)alloc((size_t)N_ATOM * C_ATOM * 4);
  float* H1     = (float*)alloc((size_t)N_ATOM * 256 * 4);
  float* H2     = (float*)alloc((size_t)N_ATOM * 256 * 4);
  bf16*  hhb    = (bf16*)alloc((size_t)N_ATOM * 256 * 2);
  float* abuf   = (float*)alloc((size_t)N_ATOM * C_TOKEN * 4);
  bf16*  wt0    = (bf16*)alloc((size_t)16384 * 2);   // <=128x128 transposed weights
  bf16*  wt1    = (bf16*)alloc((size_t)(C_ATOM * FPAD) * 2);  // largest transposed weight
  bf16*  m1T    = (bf16*)alloc((size_t)512 * 2);
  bf16*  m2T    = (bf16*)alloc((size_t)512 * 2);
  bf16*  m3T    = (bf16*)alloc((size_t)512 * 2);

  // ---- trunk single: s_proj = ln(s) @ w_s ----
  k_ln<<<(N_TOKEN + 3) / 4, 128, 0, stream>>>(s_in, 384, nullptr, s_ln, N_TOKEN);
  castT(stream, PL(55), wt1, 384, C_ATOM, 384);                      // w_s
  gemm(stream, s_ln, 384, wt1, 384, s_proj, nullptr, nullptr,
       N_TOKEN, C_ATOM, 384, C_ATOM, 0, 0);

  // ---- atom embedding: cfeat = feats @ w_feat via WMMA (K padded 389->416) ----
  k_packfeat<<<(N_ATOM * FPAD + 255) / 256, 256, 0, stream>>>(
      ref_pos, ref_charge, ref_mask, ref_elem, ref_anc, featb);
  castT(stream, PL(51), wt1, FEAT, C_ATOM, FPAD);                    // w_feat
  gemm(stream, featb, FPAD, wt1, FPAD, cfeat, nullptr, nullptr,
       N_ATOM, C_ATOM, FPAD, C_ATOM, 0, 0);
  k_embed2<<<(N_ATOM * C_ATOM + 255) / 256, 256, 0, stream>>>(
      cfeat, r_l, PL(54), s_proj, a2t, out_c, qbuf, crb);            // w_r
  k_ln<<<(N_ATOM + 3) / 4, 128, 0, stream>>>(out_c, C_ATOM, nullptr, lnc, N_ATOM);

  // ---- trunk pair: zp = ln(z) @ w_z ----
  k_ln<<<(N_TOKEN * N_TOKEN + 3) / 4, 128, 0, stream>>>(z_in, C_Z, nullptr, big0,
                                                        N_TOKEN * N_TOKEN);
  castT(stream, PL(57), wt0, C_Z, C_PAIR, C_Z);                      // w_z
  gemm(stream, big0, C_Z, wt0, C_Z, zp, nullptr, nullptr,
       N_TOKEN * N_TOKEN, C_PAIR, C_Z, C_PAIR, 0, 0);

  // ---- cq / ck = relu(c) @ w_cq / w_ck ----
  castT(stream, PL(49), wt0, C_ATOM, C_PAIR, C_ATOM);                // w_cq
  gemm(stream, crb, C_ATOM, wt0, C_ATOM, cqb, nullptr, nullptr,
       N_ATOM, C_PAIR, C_ATOM, C_PAIR, 0, 0);
  castT(stream, PL(48), wt0, C_ATOM, C_PAIR, C_ATOM);                // w_ck
  gemm(stream, crb, C_ATOM, wt0, C_ATOM, ckb, nullptr, nullptr,
       N_ATOM, C_PAIR, C_ATOM, C_PAIR, 0, 0);

  // ---- fused pair stack: p0 + 3-layer MLP entirely in LDS ----
  castT(stream, PL(45), m1T, C_PAIR, C_PAIR, 32);                    // m1
  castT(stream, PL(46), m2T, C_PAIR, C_PAIR, 32);                    // m2
  castT(stream, PL(47), m3T, C_PAIR, C_PAIR, 32);                    // m3
  k_pair_fused<<<N_ATOM, 256, 0, stream>>>(
      ref_pos, uid, a2t, zp, cqb, ckb, PL(50), PL(52), PL(56),       // w_d, w_inv, w_v
      m1T, m2T, m3T, out_p);

  // ---- pair bias for all 3 blocks in one pass over p ----
  k_bias3<<<(NPAIR + 255) / 256, 256, 0, stream>>>(
      out_p, PL(0 * 15 + 7), PL(1 * 15 + 7), PL(2 * 15 + 7),
      bq3, bq3 + (size_t)NPAIR * NH, bq3 + (size_t)2 * NPAIR * NH);

  // ---- transformer blocks ----
  for (int b = 0; b < 3; ++b) {
    int L = b * 15;  // sorted keys: ada_b, ada_b2, ada_g, ada_g2, w1, w2, w3,
                     //              w_bias, w_gate, w_skip, w_skip2, wk, wo, wq, wv
    const float* ada_b   = PL(L + 0);
    const float* ada_b2  = PL(L + 1);
    const float* ada_g   = PL(L + 2);
    const float* ada_g2  = PL(L + 3);
    const float* w1      = PL(L + 4);
    const float* w2      = PL(L + 5);
    const float* w3      = PL(L + 6);
    const float* w_gate  = PL(L + 8);
    const float* w_skip  = PL(L + 9);
    const float* w_skip2 = PL(L + 10);
    const float* wk      = PL(L + 11);
    const float* wo      = PL(L + 12);
    const float* wq      = PL(L + 13);
    const float* wv      = PL(L + 14);

    castT(stream, ada_g, wt0, C_ATOM, C_ATOM, C_ATOM);
    gemm(stream, lnc, C_ATOM, wt0, C_ATOM, G1, nullptr, nullptr, N_ATOM, C_ATOM, C_ATOM, C_ATOM, 0, 0);
    castT(stream, ada_b, wt0, C_ATOM, C_ATOM, C_ATOM);
    gemm(stream, lnc, C_ATOM, wt0, C_ATOM, G2, nullptr, nullptr, N_ATOM, C_ATOM, C_ATOM, C_ATOM, 0, 0);
    k_adaln<<<(N_ATOM + 3) / 4, 128, 0, stream>>>(qbuf, G1, G2, bbf);

    castT(stream, wq, wt0, C_ATOM, C_ATOM, C_ATOM);
    gemm(stream, bbf, C_ATOM, wt0, C_ATOM, nullptr, qhb, nullptr, N_ATOM, C_ATOM, C_ATOM, 0, C_ATOM, 0);
    castT(stream, wk, wt0, C_ATOM, C_ATOM, C_ATOM);
    gemm(stream, bbf, C_ATOM, wt0, C_ATOM, nullptr, khb, nullptr, N_ATOM, C_ATOM, C_ATOM, 0, C_ATOM, 0);
    castT(stream, wv, wt0, C_ATOM, C_ATOM, C_ATOM);
    gemm(stream, bbf, C_ATOM, wt0, C_ATOM, nullptr, vhb, nullptr, N_ATOM, C_ATOM, C_ATOM, 0, C_ATOM, 0);

    k_attn<<<NBQ, 256, 0, stream>>>(qhb, khb, vhb, bq3 + (size_t)b * NPAIR * NH, obuf);

    castT(stream, w_gate, wt0, C_ATOM, C_ATOM, C_ATOM);
    gemm(stream, bbf, C_ATOM, wt0, C_ATOM, G1, nullptr, nullptr, N_ATOM, C_ATOM, C_ATOM, C_ATOM, 0, 0);
    k_gate<<<(N_ATOM * C_ATOM + 255) / 256, 256, 0, stream>>>(G1, obuf, bbf, N_ATOM * C_ATOM);
    castT(stream, wo, wt0, C_ATOM, C_ATOM, C_ATOM);
    gemm(stream, bbf, C_ATOM, wt0, C_ATOM, G1, nullptr, nullptr, N_ATOM, C_ATOM, C_ATOM, C_ATOM, 0, 0);
    castT(stream, w_skip, wt0, C_ATOM, C_ATOM, C_ATOM);
    gemm(stream, lnc, C_ATOM, wt0, C_ATOM, G2, nullptr, nullptr, N_ATOM, C_ATOM, C_ATOM, C_ATOM, 0, 0);
    k_skipadd<<<(N_ATOM * C_ATOM + 255) / 256, 256, 0, stream>>>(qbuf, G2, G1, N_ATOM * C_ATOM);

    castT(stream, ada_g2, wt0, C_ATOM, C_ATOM, C_ATOM);
    gemm(stream, lnc, C_ATOM, wt0, C_ATOM, G1, nullptr, nullptr, N_ATOM, C_ATOM, C_ATOM, C_ATOM, 0, 0);
    castT(stream, ada_b2, wt0, C_ATOM, C_ATOM, C_ATOM);
    gemm(stream, lnc, C_ATOM, wt0, C_ATOM, G2, nullptr, nullptr, N_ATOM, C_ATOM, C_ATOM, C_ATOM, 0, 0);
    k_adaln<<<(N_ATOM + 3) / 4, 128, 0, stream>>>(qbuf, G1, G2, bbf);

    castT(stream, w1, wt1, C_ATOM, 256, C_ATOM);
    gemm(stream, bbf, C_ATOM, wt1, C_ATOM, H1, nullptr, nullptr, N_ATOM, 256, C_ATOM, 256, 0, 0);
    castT(stream, w2, wt1, C_ATOM, 256, C_ATOM);
    gemm(stream, bbf, C_ATOM, wt1, C_ATOM, H2, nullptr, nullptr, N_ATOM, 256, C_ATOM, 256, 0, 0);
    k_silu2<<<(N_ATOM * 256 + 255) / 256, 256, 0, stream>>>(H1, H2, hhb, N_ATOM * 256);
    castT(stream, w3, wt1, 256, C_ATOM, 256);
    gemm(stream, hhb, 256, wt1, 256, obuf, nullptr, nullptr, N_ATOM, C_ATOM, 256, C_ATOM, 0, 0);
    castT(stream, w_skip2, wt0, C_ATOM, C_ATOM, C_ATOM);
    gemm(stream, lnc, C_ATOM, wt0, C_ATOM, G1, nullptr, nullptr, N_ATOM, C_ATOM, C_ATOM, C_ATOM, 0, 0);
    k_skipadd<<<(N_ATOM * C_ATOM + 255) / 256, 256, 0, stream>>>(qbuf, G1, obuf, N_ATOM * C_ATOM);
  }

  // ---- output head: a = relu(q @ w_out), token mean ----
  k_cast<<<(N_ATOM * C_ATOM + 255) / 256, 256, 0, stream>>>(qbuf, bbf, N_ATOM * C_ATOM);
  castT(stream, PL(53), wt1, C_ATOM, C_TOKEN, C_ATOM);               // w_out
  gemm(stream, bbf, C_ATOM, wt1, C_ATOM, abuf, nullptr, nullptr,
       N_ATOM, C_TOKEN, C_ATOM, C_TOKEN, 0, 1);
  k_mean<<<(N_TOKEN * C_TOKEN + 255) / 256, 256, 0, stream>>>(abuf, out_atok);
  hipMemcpyAsync(out_q, qbuf, (size_t)N_ATOM * C_ATOM * sizeof(float),
                 hipMemcpyDeviceToDevice, stream);
}
